// RSNN3_52613349376371
// MI455X (gfx1250) — compile-verified, matrix operand order checked by reading, not run on hardware
//
#include <hip/hip_runtime.h>

typedef __attribute__((ext_vector_type(16))) _Float16 v16h;
typedef __attribute__((ext_vector_type(8)))  float    v8f;

#define T_STEPS 1000
#define BATCH   128
#define NH      200
#define NIN     3
#define BCHUNK  16
#define NTILES  13          // 13 * 16 = 208 padded neurons (row 200 = W2 readout row)
#define KCH     7           // 7 * 32 = 224 padded K
#define SPK_STRIDE 232      // halves per LDS row = 116 dwords -> conflict-free A loads
#define SPK_BUF  (BCHUNK * SPK_STRIDE)
#define NTHREADS (NTILES*32)

union H16 { _Float16 f[16]; unsigned int u[8]; v16h h; };

__global__ __launch_bounds__(NTHREADS) void rsnn_kernel(
    const float* __restrict__ inp, const float* __restrict__ W1,
    const float* __restrict__ b1,  const float* __restrict__ Wr,
    const float* __restrict__ br,  const float* __restrict__ W2,
    const float* __restrict__ b2,  const float* __restrict__ beta,
    float* __restrict__ out_cur2,  float* __restrict__ out_spk)
{
    // double-buffered spike matrix (f16, padded) -> only ONE barrier per step
    __shared__ _Float16 lds_spk[2 * SPK_BUF];
    __shared__ float    xbuf[2][BCHUNK * NIN];

    const int tid   = threadIdx.x;
    const int wid   = tid >> 5;
    const int lane  = tid & 31;
    const int b0    = blockIdx.x * BCHUNK;          // 8 blocks x 16 batch rows
    const int nloc  = (wid << 4) + (lane & 15);     // 0..207 ; 200 == readout row
    const int klo   = (lane >= 16) ? 8 : 0;         // K offset (halves) per WMMA A/B layout
    const int mbase = (lane >= 16) ? 8 : 0;         // M offset of this lane's C/D rows

    // zero both spike buffers (padded columns must stay zero forever)
    for (int i = tid; i < 2 * SPK_BUF; i += NTHREADS) lds_spk[i] = (_Float16)0.0f;

    // per-lane neuron constants (N is fixed per lane in the C/D layout)
    float beta_n = 0.f, br_n = 0.f, b1_n = 0.f, w10 = 0.f, w11 = 0.f, w12 = 0.f;
    if (nloc < NH) {
        beta_n = fminf(fmaxf(beta[nloc], 0.0f), 1.0f);
        br_n = br[nloc]; b1_n = b1[nloc];
        w10 = W1[nloc*NIN+0]; w11 = W1[nloc*NIN+1]; w12 = W1[nloc*NIN+2];
    }
    const float b2v = b2[0];

    // Preload time-invariant B fragments into registers: B^T[n][k] = Wr[n][k];
    // row 200 carries W2 so the matmul emits the readout for free.
    v16h bfrag[KCH];
#pragma unroll
    for (int kc = 0; kc < KCH; ++kc) {
        H16 ub;
#pragma unroll
        for (int j = 0; j < 16; ++j) {
            int v = j >> 1, p = j & 1;
            int kl = ((v & 4) ? 16 : 0) + ((v & 3) << 1) + klo + p;
            int k  = kc * 32 + kl;
            float val = 0.0f;
            if (k < NH) {
                if (nloc < NH)       val = Wr[nloc*NH + k];
                else if (nloc == NH) val = W2[k];
            }
            ub.f[j] = (_Float16)val;
        }
        bfrag[kc] = ub.h;
    }

    float memv[8];
#pragma unroll
    for (int r = 0; r < 8; ++r) memv[r] = 0.0f;

    const unsigned int* spk_dw = (const unsigned int*)lds_spk;
    const int arow = (lane & 15) * (SPK_STRIDE / 2);   // dwords within a buffer
    const int aklo = klo >> 1;                         // dwords

    for (int t = 0; t < T_STEPS; ++t) {
        const int rb = t & 1;            // read spikes(t-1) from this buffer
        const int wb = rb ^ 1;           // write spikes(t) to the other buffer

        if (tid < BCHUNK*NIN) xbuf[rb][tid] = inp[(size_t)(t*BATCH + b0)*NIN + tid];
        __syncthreads();   // spikes(t-1) writes + x(t) visible; sole barrier of the step

        v8f acc = {};
        const unsigned int* bufp = spk_dw + rb * (SPK_BUF/2) + arow;
#pragma unroll
        for (int kc = 0; kc < KCH; ++kc) {
            H16 ua;
            const unsigned int* rowp = bufp + kc*16;
#pragma unroll
            for (int j = 0; j < 4; ++j) { ua.u[j] = rowp[aklo + j]; ua.u[4+j] = rowp[8 + aklo + j]; }
            acc = __builtin_amdgcn_wmma_f32_16x16x32_f16(
                false, ua.h, false, bfrag[kc], (short)0, acc, false, false);
        }

        if (nloc < NH) {
            _Float16* spk_w = lds_spk + wb * SPK_BUF;
#pragma unroll
            for (int r = 0; r < 8; ++r) {
                int m = mbase + r;
                float cur1 = w10*xbuf[rb][m*NIN+0] + w11*xbuf[rb][m*NIN+1]
                           + w12*xbuf[rb][m*NIN+2] + b1_n;
                float base = beta_n*memv[r] + cur1 + acc[r] + br_n;
                float mem_new = (memv[r] - 1.0f > 0.0f) ? 0.0f : base;   // zero reset
                float spk = (mem_new - 1.0f > 0.0f) ? 1.0f : 0.0f;
                memv[r] = mem_new;
                spk_w[m*SPK_STRIDE + nloc] = (_Float16)spk;              // for step t+1
                out_spk[((size_t)t*BATCH + b0 + m)*NH + nloc] = spk;
            }
        } else if (nloc == NH && t > 0) {
            // acc here = spk_new(t-1) . W2  -> cur2(t-1)
#pragma unroll
            for (int r = 0; r < 8; ++r)
                out_cur2[(size_t)(t-1)*BATCH + b0 + mbase + r] = acc[r] + b2v;
        }
    }

    // tail: cur2 for t = 999 lives in buffer (T_STEPS & 1)
    __syncthreads();
    if (wid == NTILES - 1) {      // wave-uniform branch keeps EXEC all-ones for WMMA
        v8f acc = {};
        const unsigned int* bufp = spk_dw + (T_STEPS & 1) * (SPK_BUF/2) + arow;
#pragma unroll
        for (int kc = 0; kc < KCH; ++kc) {
            H16 ua;
            const unsigned int* rowp = bufp + kc*16;
#pragma unroll
            for (int j = 0; j < 4; ++j) { ua.u[j] = rowp[aklo + j]; ua.u[4+j] = rowp[8 + aklo + j]; }
            acc = __builtin_amdgcn_wmma_f32_16x16x32_f16(
                false, ua.h, false, bfrag[kc], (short)0, acc, false, false);
        }
        if (nloc == NH) {
#pragma unroll
            for (int r = 0; r < 8; ++r)
                out_cur2[(size_t)(T_STEPS-1)*BATCH + b0 + mbase + r] = acc[r] + b2v;
        }
    }
}

extern "C" void kernel_launch(void* const* d_in, const int* in_sizes, int n_in,
                              void* d_out, int out_size, void* d_ws, size_t ws_size,
                              hipStream_t stream) {
    const float* inp  = (const float*)d_in[0];
    const float* W1   = (const float*)d_in[1];
    const float* b1   = (const float*)d_in[2];
    const float* Wr   = (const float*)d_in[3];
    const float* br   = (const float*)d_in[4];
    const float* W2   = (const float*)d_in[5];
    const float* b2   = (const float*)d_in[6];
    const float* beta = (const float*)d_in[7];

    float* out_cur2 = (float*)d_out;                              // [1000,128,1]
    float* out_spk  = out_cur2 + (size_t)T_STEPS * BATCH;         // [1000,128,200]

    rsnn_kernel<<<dim3(BATCH / BCHUNK), dim3(NTHREADS), 0, stream>>>(
        inp, W1, b1, Wr, br, W2, b2, beta, out_cur2, out_spk);
}
